// ScatteringAwareAttention_89996744720959
// MI455X (gfx1250) — compile-verified, matrix operand order checked by reading, not run on hardware
//
#include <hip/hip_runtime.h>

// ---------------------------------------------------------------------------
// Types for CDNA5 WMMA (wave32, 16x16x32 bf16 -> f32)
// ---------------------------------------------------------------------------
typedef __attribute__((ext_vector_type(8)))  __bf16 v8bf;
typedef __attribute__((ext_vector_type(16))) __bf16 v16bf;
typedef __attribute__((ext_vector_type(8)))  float  v8f;

#define WMMA_BF16(a, b, c) \
    __builtin_amdgcn_wmma_f32_16x16x32_bf16(false, (a), false, (b), (short)0, (c), false, false)

#define ASYNC_B128(ldsOff, gaddr)                                              \
    asm volatile("global_load_async_to_lds_b128 %0, %1, off"                   \
                 :: "v"(ldsOff), "v"((unsigned long long)(uintptr_t)(gaddr))   \
                 : "memory")
#define WAIT_ASYNC() asm volatile("s_wait_asynccnt 0x0" ::: "memory")

// Problem constants
static constexpr int B_  = 2;
static constexpr int S_  = 2048;
static constexpr int D_  = 1024;
static constexpr int H_  = 16;
static constexpr int HD_ = 64;
static constexpr int M_  = B_ * S_;   // 4096 rows

// ---------------------------------------------------------------------------
// Fragment loader: per-lane 16 bf16 = two contiguous 16B chunks at p and p+16.
// A-frag: lane holds row M=lane&15; K = (half?8:0)+0..7 and 16+(half?8:0)+0..7.
// B-frag: identical addressing against a [N][K]-major matrix (lane holds col N).
// ---------------------------------------------------------------------------
__device__ __forceinline__ v16bf load_frag16(const __bf16* p) {
    v8bf lo = *(const v8bf*)p;
    v8bf hi = *(const v8bf*)(p + 16);
    v16bf r;
#pragma unroll
    for (int i = 0; i < 8; ++i) { r[i] = lo[i]; r[i + 8] = hi[i]; }
    return r;
}

__device__ __forceinline__ float rmax16(float v) {
    v = fmaxf(v, __shfl_xor(v, 1, 32));
    v = fmaxf(v, __shfl_xor(v, 2, 32));
    v = fmaxf(v, __shfl_xor(v, 4, 32));
    v = fmaxf(v, __shfl_xor(v, 8, 32));
    return v;   // reduced within each 16-lane half
}
__device__ __forceinline__ float rsum16(float v) {
    v += __shfl_xor(v, 1, 32);
    v += __shfl_xor(v, 2, 32);
    v += __shfl_xor(v, 4, 32);
    v += __shfl_xor(v, 8, 32);
    return v;
}

// ---------------------------------------------------------------------------
// Elementwise f32 -> bf16
// ---------------------------------------------------------------------------
__global__ __launch_bounds__(256) void k_f32_to_bf16(const float* __restrict__ in,
                                                     __bf16* __restrict__ out, int n) {
    int i = blockIdx.x * 256 + threadIdx.x;
    if (i < n) out[i] = (__bf16)in[i];
}

// W[k][n] f32 (1024x1024) -> Wt[n][k] bf16
__global__ __launch_bounds__(256) void k_transpose_w(const float* __restrict__ W,
                                                     __bf16* __restrict__ Wt) {
    int idx = blockIdx.x * 256 + threadIdx.x;      // 1M threads
    int n = idx & (D_ - 1);
    int k = idx >> 10;
    Wt[(size_t)n * D_ + k] = (__bf16)W[(size_t)k * D_ + n];
}

// vb[b*S+s][h*64+d] bf16 -> vT[((b*H+h)*64+d)][s] bf16
__global__ __launch_bounds__(256) void k_transpose_v(const __bf16* __restrict__ vb,
                                                     __bf16* __restrict__ vT) {
    int idx = blockIdx.x * 256 + threadIdx.x;      // 4M threads, idx == vT linear index
    int s = idx & (S_ - 1);
    int d = (idx >> 11) & (HD_ - 1);
    int h = (idx >> 17) & (H_ - 1);
    int b = idx >> 21;
    vT[idx] = vb[(size_t)(b * S_ + s) * D_ + h * HD_ + d];
}

// keep[b*32+q] = (max |G| over 64-row block) >= 0.99
__global__ __launch_bounds__(64) void k_keep(const float* __restrict__ G,
                                             int* __restrict__ keep) {
    __shared__ float sm[64];
    int blk = blockIdx.x;                  // 0..63  (b*32 + qblock)
    int b = blk >> 5, q = blk & 31;
    sm[threadIdx.x] = fabsf(G[(size_t)b * S_ + q * 64 + threadIdx.x]);
    __syncthreads();
    if (threadIdx.x == 0) {
        float m = 0.f;
        for (int i = 0; i < 64; ++i) m = fmaxf(m, sm[i]);
        keep[blk] = (m >= 0.99f) ? 1 : 0;
    }
}

// ---------------------------------------------------------------------------
// GEMM: C[M][N] = A[M][K] @ Bt[N][K]^T + bias.
// Block = 32 rows x 256 cols, 8 waves as 2x4 grid of 16x64 tiles.
// A (4 KB) and B (32 KB) k-chunks staged into padded LDS with gfx1250
// async-to-LDS DMA, double-buffered.  All 10 fragment loads of a chunk are
// issued before the 8-WMMA chain so a single DScnt wait covers the clause.
// Rows padded to 72 elements (144 B) -> conflict-free 16-lane ds_load_b128.
// ---------------------------------------------------------------------------
__global__ __launch_bounds__(256) void k_gemm_wmma(const __bf16* __restrict__ A,
                                                   const __bf16* __restrict__ Bt,
                                                   const float* __restrict__ bias,
                                                   float* Cf, __bf16* Cb,
                                                   int M, int N, int K) {
    __shared__ __align__(16) __bf16 sA[2][32][72];    //  9 KB
    __shared__ __align__(16) __bf16 sB[2][256][72];   // 72 KB

    const int tid  = threadIdx.x;
    const int wave = tid >> 5;
    const int lane = tid & 31;
    const int half = lane >> 4, l15 = lane & 15;
    const int wm = wave >> 2, wn = wave & 3;          // 2 x 4 wave grid

    const int nb_per_row = N >> 8;                    // column-blocks of 256
    const int mbase = (blockIdx.x / nb_per_row) << 5;
    const int nbase = (blockIdx.x % nb_per_row) << 8;
    const int n0 = nbase + wn * 64;

    const unsigned sA0 = (unsigned)(uintptr_t)&sA[0][0][0];
    const unsigned sA1 = (unsigned)(uintptr_t)&sA[1][0][0];
    const unsigned sB0 = (unsigned)(uintptr_t)&sB[0][0][0];
    const unsigned sB1 = (unsigned)(uintptr_t)&sB[1][0][0];

    // A copy: 32 rows x 128B = 256 x 16B chunks, 1 per thread
    const int arow = tid >> 3, ain = (tid & 7) * 16;
    const char* ag = (const char*)(A + (size_t)(mbase + arow) * K) + ain;
    const unsigned aLds = arow * 144 + ain;

    // B copy: 256 rows x 128B = 2048 x 16B chunks, 8 per thread
    auto issue_chunk = [&](int ch, unsigned sAb, unsigned sBb) {
        const size_t kByte = (size_t)ch * 128;        // 64 k * 2B
        ASYNC_B128(sAb + aLds, ag + kByte);
#pragma unroll
        for (int j = 0; j < 8; ++j) {
            int cidx = tid + j * 256;
            int row = cidx >> 3, inrow = (cidx & 7) * 16;
            const char* g = (const char*)(Bt + (size_t)(nbase + row) * K) + kByte + inrow;
            ASYNC_B128(sBb + row * 144 + inrow, g);
        }
    };

    issue_chunk(0, sA0, sB0);
    WAIT_ASYNC();
    __syncthreads();

    v8f c[4] = {};
    const int nchunks = K >> 6;                       // 16
    for (int ch = 0; ch < nchunks; ++ch) {
        if (ch + 1 < nchunks)
            issue_chunk(ch + 1, ((ch + 1) & 1) ? sA1 : sA0, ((ch + 1) & 1) ? sB1 : sB0);

        const int buf = ch & 1;
        const __bf16* aRow = &sA[buf][wm * 16 + l15][half * 8];

        // issue ALL fragment loads for this chunk first (one LDS round trip)
        v16bf a0 = load_frag16(aRow);
        v16bf a1 = load_frag16(aRow + 32);
        v16bf b0[4], b1[4];
#pragma unroll
        for (int n = 0; n < 4; ++n) {
            const __bf16* bRow = &sB[buf][wn * 64 + n * 16 + l15][half * 8];
            b0[n] = load_frag16(bRow);
            b1[n] = load_frag16(bRow + 32);
        }
        // back-to-back WMMA chain
#pragma unroll
        for (int n = 0; n < 4; ++n) {
            c[n] = WMMA_BF16(a0, b0[n], c[n]);
            c[n] = WMMA_BF16(a1, b1[n], c[n]);
        }

        WAIT_ASYNC();          // chunk c+1 DMA landed
        __syncthreads();       // all waves done reading chunk c
    }

#pragma unroll
    for (int n = 0; n < 4; ++n) {
        float bv = bias ? bias[n0 + n * 16 + l15] : 0.f;
#pragma unroll
        for (int i = 0; i < 8; ++i) {
            float v = c[n][i] + bv;
            size_t idx = (size_t)(mbase + wm * 16 + half * 8 + i) * N + n0 + n * 16 + l15;
            if (Cf) Cf[idx] = v;
            if (Cb) Cb[idx] = (__bf16)v;
        }
    }
}

// ---------------------------------------------------------------------------
// Flash attention: block = 8 query tiles (16 rows each) of one (b,h); all
// waves share the K/V stream, so 32-key K (4 KB) and V (4 KB) chunks are
// async-staged into double-buffered LDS by the whole block.  K and V frags
// are all loaded up front; V loads overlap the softmax VALU work.
// Waves whose 64-query keep-block is dropped idle through the barrier loop.
// ---------------------------------------------------------------------------
__global__ __launch_bounds__(256) void k_attention(const __bf16* __restrict__ qb,
                                                   const __bf16* __restrict__ kb,
                                                   const __bf16* __restrict__ vT,
                                                   const int* __restrict__ keep,
                                                   __bf16* __restrict__ aob) {
    __shared__ __align__(16) __bf16 sK[2][32][72];     // keys x d   (9 KB)
    __shared__ __align__(16) __bf16 sV[2][64][40];     // d x keys   (10 KB)
    __shared__ __align__(16) __bf16 plds[8][16][32];   // per-wave P (8 KB)

    const int tid  = threadIdx.x;
    const int wave = tid >> 5;
    const int lane = tid & 31;
    const int half = lane >> 4, l15 = lane & 15;

    const int t0 = blockIdx.x * 8;          // first tile of block
    const int h  = (t0 >> 7) & 15;          // block-uniform
    const int b  = t0 >> 11;                // block-uniform
    const int qt = (t0 & 127) + wave;       // per-wave query tile
    const int row0 = qt << 4;
    const int col  = h * HD_;

    const bool kept = keep[b * 32 + (row0 >> 6)] != 0;

    const size_t krow  = (size_t)b * S_;
    const size_t vbase = (size_t)((b * H_ + h) * HD_);

    // K copy: 32 rows x 128B = 256 x 16B, 1 per thread
    const int krow_c = tid >> 3, kin_c = (tid & 7) * 16;
    const char* kg = (const char*)(kb + (krow + krow_c) * D_ + col) + kin_c;
    const unsigned kLds = krow_c * 144 + kin_c;
    // V copy: 64 rows x 64B = 256 x 16B, 1 per thread
    const int vrow_c = tid >> 2, vin_c = (tid & 3) * 16;
    const char* vg = (const char*)(vT + (vbase + vrow_c) * S_) + vin_c;
    const unsigned vLds = vrow_c * 80 + vin_c;

    const unsigned sK0 = (unsigned)(uintptr_t)&sK[0][0][0];
    const unsigned sK1 = (unsigned)(uintptr_t)&sK[1][0][0];
    const unsigned sV0 = (unsigned)(uintptr_t)&sV[0][0][0];
    const unsigned sV1 = (unsigned)(uintptr_t)&sV[1][0][0];

    auto issue_kv = [&](int key0, int buf) {
        ASYNC_B128((buf ? sK1 : sK0) + kLds, kg + (size_t)key0 * D_ * 2);
        ASYNC_B128((buf ? sV1 : sV0) + vLds, vg + (size_t)key0 * 2);
    };

    // Q fragments (two 16x32 k-slices over head dim), loaded once from global
    v16bf qf[2];
#pragma unroll
    for (int ks = 0; ks < 2; ++ks)
        qf[ks] = load_frag16(qb + (size_t)(b * S_ + row0 + l15) * D_ + col + ks * 32 + half * 8);

    float m_i[8], l_i[8], alpha[8];
#pragma unroll
    for (int i = 0; i < 8; ++i) { m_i[i] = -1e30f; l_i[i] = 0.f; }
    v8f acc[4] = {};

    issue_kv(0, 0);
    WAIT_ASYNC();
    __syncthreads();

    const int nsteps = S_ / 32;             // 64
    for (int ch = 0; ch < nsteps; ++ch) {
        if (ch + 1 < nsteps) issue_kv((ch + 1) * 32, (ch + 1) & 1);
        const int buf = ch & 1;

        if (kept) {
            // ---- issue all K and V fragment loads up front ----
            v16bf kf0a = load_frag16(&sK[buf][l15][half * 8]);
            v16bf kf0b = load_frag16(&sK[buf][l15][32 + half * 8]);
            v16bf kf1a = load_frag16(&sK[buf][16 + l15][half * 8]);
            v16bf kf1b = load_frag16(&sK[buf][16 + l15][32 + half * 8]);
            v16bf vf[4];
#pragma unroll
            for (int n = 0; n < 4; ++n)
                vf[n] = load_frag16(&sV[buf][n * 16 + l15][half * 8]);

            // ---- scores: two 16x16 tiles over the 32 staged keys ----
            v8f s0 = {}, s1 = {};
            s0 = WMMA_BF16(qf[0], kf0a, s0);
            s0 = WMMA_BF16(qf[1], kf0b, s0);
            s1 = WMMA_BF16(qf[0], kf1a, s1);
            s1 = WMMA_BF16(qf[1], kf1b, s1);

            // ---- online softmax (row stats align with C-frag row mapping) ----
#pragma unroll
            for (int i = 0; i < 8; ++i) {
                float sa = s0[i] * 0.125f;          // 1/sqrt(64)
                float sb = s1[i] * 0.125f;
                float mx = rmax16(fmaxf(sa, sb));
                float mt = fmaxf(m_i[i], mx);
                float al = __expf(m_i[i] - mt);
                float p0 = __expf(sa - mt);
                float p1 = __expf(sb - mt);
                m_i[i] = mt;
                alpha[i] = al;
                l_i[i] = l_i[i] * al + rsum16(p0 + p1);
                s0[i] = p0;
                s1[i] = p1;
            }
#pragma unroll
            for (int n = 0; n < 4; ++n)
#pragma unroll
                for (int i = 0; i < 8; ++i) acc[n][i] *= alpha[i];

            // ---- P tile: C-layout -> LDS -> A-layout bf16 fragment ----
#pragma unroll
            for (int i = 0; i < 8; ++i) {
                int r = half * 8 + i;
                plds[wave][r][l15]      = (__bf16)s0[i];
                plds[wave][r][16 + l15] = (__bf16)s1[i];
            }
            asm volatile("s_wait_dscnt 0x0" ::: "memory");
            __builtin_amdgcn_wave_barrier();
            v16bf pf = load_frag16(&plds[wave][l15][half * 8]);

            // ---- accumulate P @ V over 4 output tiles (d = 0..63) ----
#pragma unroll
            for (int n = 0; n < 4; ++n)
                acc[n] = WMMA_BF16(pf, vf[n], acc[n]);
        }

        WAIT_ASYNC();          // next chunk's DMA landed
        __syncthreads();       // all waves done with this chunk's LDS
    }

    // ---- finalize: divide by l (or zero for dropped blocks), store bf16 ----
#pragma unroll
    for (int i = 0; i < 8; ++i) {
        float inv = kept ? (1.0f / l_i[i]) : 0.f;
        size_t rbase = (size_t)(b * S_ + row0 + half * 8 + i) * D_ + col;
#pragma unroll
        for (int n = 0; n < 4; ++n)
            aob[rbase + n * 16 + l15] = kept ? (__bf16)(acc[n][i] * inv) : (__bf16)0.0f;
    }
}

// ---------------------------------------------------------------------------
// Host-side orchestration
// ---------------------------------------------------------------------------
extern "C" void kernel_launch(void* const* d_in, const int* in_sizes, int n_in,
                              void* d_out, int out_size, void* d_ws, size_t ws_size,
                              hipStream_t stream) {
    (void)in_sizes; (void)n_in; (void)out_size; (void)ws_size;

    const float* x  = (const float*)d_in[0];
    const float* G  = (const float*)d_in[1];
    const float* Wq = (const float*)d_in[2];
    const float* bq = (const float*)d_in[3];
    const float* Wk = (const float*)d_in[4];
    const float* bk = (const float*)d_in[5];
    const float* Wv = (const float*)d_in[6];
    const float* bv = (const float*)d_in[7];
    const float* Wo = (const float*)d_in[8];
    const float* bo = (const float*)d_in[9];
    float* out = (float*)d_out;

    char* ws = (char*)d_ws;
    const size_t MB = 1u << 20;
    __bf16* xb  = (__bf16*)(ws + 0 * MB);      // 8 MB   x in bf16
    __bf16* wqt = (__bf16*)(ws + 8 * MB);      // 2 MB   Wq^T bf16
    __bf16* wkt = (__bf16*)(ws + 10 * MB);
    __bf16* wvt = (__bf16*)(ws + 12 * MB);
    __bf16* wot = (__bf16*)(ws + 14 * MB);
    __bf16* qb  = (__bf16*)(ws + 16 * MB);     // 8 MB   Q bf16
    __bf16* kbf = (__bf16*)(ws + 24 * MB);     // 8 MB   K bf16
    __bf16* vb  = (__bf16*)(ws + 32 * MB);     // 8 MB   V bf16 row-major
    __bf16* vT  = (__bf16*)(ws + 40 * MB);     // 8 MB   V^T per head
    __bf16* aob = (__bf16*)(ws + 48 * MB);     // 8 MB   attention out bf16
    int*    keep = (int*)(ws + 56 * MB);       // 256 B

    const int nX = M_ * D_;                    // 4,194,304
    k_f32_to_bf16<<<nX / 256, 256, 0, stream>>>(x, xb, nX);

    k_transpose_w<<<(D_ * D_) / 256, 256, 0, stream>>>(Wq, wqt);
    k_transpose_w<<<(D_ * D_) / 256, 256, 0, stream>>>(Wk, wkt);
    k_transpose_w<<<(D_ * D_) / 256, 256, 0, stream>>>(Wv, wvt);
    k_transpose_w<<<(D_ * D_) / 256, 256, 0, stream>>>(Wo, wot);

    k_keep<<<B_ * 32, 64, 0, stream>>>(G, keep);

    // QKV projections: 4096x1024x1024, bf16 outputs
    const int gemm_blocks = (M_ / 32) * (D_ / 256);        // 512
    k_gemm_wmma<<<gemm_blocks, 256, 0, stream>>>(xb, wqt, bq, nullptr, qb,  M_, D_, D_);
    k_gemm_wmma<<<gemm_blocks, 256, 0, stream>>>(xb, wkt, bk, nullptr, kbf, M_, D_, D_);
    k_gemm_wmma<<<gemm_blocks, 256, 0, stream>>>(xb, wvt, bv, nullptr, vb,  M_, D_, D_);

    k_transpose_v<<<(B_ * H_ * HD_ * S_) / 256, 256, 0, stream>>>(vb, vT);

    // Attention: 4096 (b,h,qtile) waves, 8 waves/block
    k_attention<<<(B_ * H_ * (S_ / 16)) / 8, 256, 0, stream>>>(qb, kbf, vT, keep, aob);

    // Output projection -> f32 d_out
    k_gemm_wmma<<<gemm_blocks, 256, 0, stream>>>(aob, wot, bo, out, nullptr, M_, D_, D_);
}
